// DecoderRNN_58171037057784
// MI455X (gfx1250) — matrix operand order repaired and checked by reading, not compiled, not run on hardware
//
#include <hip/hip_runtime.h>
#include <math.h>

// Problem sizes (fixed by the reference)
#define SS 1024      // encoder sequence length
#define BB 64        // batch
#define HH 1024      // hidden
#define VV 50257     // vocab
#define FOURH 4096
#define H2 2048

typedef __attribute__((ext_vector_type(16))) __bf16 v16bf;
typedef __attribute__((ext_vector_type(8)))  __bf16 v8bf;
typedef __attribute__((ext_vector_type(8)))  float  v8f;
typedef __attribute__((ext_vector_type(8)))  float  v8fv;

// ---------- helpers ----------

__device__ inline __bf16 f2bf(float f) { return (__bf16)f; }  // HW cvt, RTNE

__device__ inline float sigm(float x) { return 1.f / (1.f + expf(-x)); }

__device__ inline v16bf cat8(v8bf a, v8bf b) {
  return __builtin_shufflevector(a, b, 0, 1, 2, 3, 4, 5, 6, 7,
                                       8, 9, 10, 11, 12, 13, 14, 15);
}

// A tile (16x32 bf16, MxK): rk points at A[row] + k0 (row = m0 + (lane&15))
// ISA layout: lane<16 -> elems 0..7 = K0..7, elems 8..15 = K16..23
//             lane>=16 -> elems 0..7 = K8..15, elems 8..15 = K24..31
__device__ inline v16bf load_a_row(const __bf16* rk, int hi) {
  const __bf16* p = rk + hi * 8;
  v8bf lo8 = __builtin_bit_cast(v8bf, *reinterpret_cast<const uint4*>(p));
  v8bf hi8 = __builtin_bit_cast(v8bf, *reinterpret_cast<const uint4*>(p + 16));
  return cat8(lo8, hi8);
}

// B tile (32x16 bf16, KxN) from row-major W[j,k] (column n = W row j):
// rk points at W[j] + k0 ; lane<16 holds K0..7,K16..23, lane>=16 holds K8..15,K24..31
// f32 source converted in-register via packed v_cvt (no extra memory pass).
__device__ inline v16bf load_b_row(const float* rk, int hi) {
  const float* p = rk + hi * 8;
  v8bf c0 = __builtin_convertvector(*reinterpret_cast<const v8fv*>(p), v8bf);
  v8bf c1 = __builtin_convertvector(*reinterpret_cast<const v8fv*>(p + 16), v8bf);
  return cat8(c0, c1);
}

__device__ inline v8f wmma_bf16(v16bf a, v16bf b, v8f c) {
  return __builtin_amdgcn_wmma_f32_16x16x32_bf16(
      false, a, false, b, (short)0, c, false, false);
}

// ---------- kernels ----------

// embedding gather + bf16 conversion of x and h0
__global__ void __launch_bounds__(256)
k_prep(const int* seq, const float* embed, const float* h0,
       __bf16* xbf, __bf16* h0bf) {
  int b = blockIdx.x;
  int tok = seq[b];
  for (int i = threadIdx.x; i < HH; i += 256) {
    xbf [(size_t)b * HH + i] = f2bf(embed[(size_t)tok * HH + i]);
    h0bf[(size_t)b * HH + i] = f2bf(h0[(size_t)b * HH + i]);
  }
}

// u2[h] = sum_j v[j] * Wa[j, H + h]
__global__ void __launch_bounds__(256)
k_u2(const float* Wa, const float* vvec, float* u2) {
  int h = blockIdx.x * 256 + threadIdx.x;   // grid 4 x 256
  float acc = 0.f;
  for (int j = 0; j < HH; ++j)
    acc += vvec[j] * Wa[(size_t)j * H2 + HH + h];
  u2[h] = acc;
}

// gates[b, n] = sum_k x[b,k] Wih[n,k] + h0[b,k] Whh[n,k]   (biases later)
// one wave per 16-wide N strip; 4 M-tiles cover B=64
__global__ void __launch_bounds__(32)
k_gates(const __bf16* xbf, const __bf16* h0bf,
        const float* Wih, const float* Whh, float* gates) {
  const int lane = threadIdx.x & 31, lo = lane & 15, hi = lane >> 4;
  const int n0 = blockIdx.x * 16;
  v8f acc[4] = {};
  const float* wih_row = Wih + (size_t)(n0 + lo) * HH;
  const float* whh_row = Whh + (size_t)(n0 + lo) * HH;
  for (int k0 = 0; k0 < HH; k0 += 32) {
    v16bf b1 = load_b_row(wih_row + k0, hi);
    v16bf b2 = load_b_row(whh_row + k0, hi);
#pragma unroll
    for (int t = 0; t < 4; ++t) {
      v16bf a1 = load_a_row(xbf  + (size_t)(t * 16 + lo) * HH + k0, hi);
      acc[t] = wmma_bf16(a1, b1, acc[t]);
      v16bf a2 = load_a_row(h0bf + (size_t)(t * 16 + lo) * HH + k0, hi);
      acc[t] = wmma_bf16(a2, b2, acc[t]);
    }
  }
#pragma unroll
  for (int t = 0; t < 4; ++t)
#pragma unroll
    for (int r = 0; r < 8; ++r)
      gates[(size_t)(t * 16 + hi * 8 + r) * FOURH + n0 + lo] = acc[t][r];
}

// LSTM pointwise: PyTorch gate order i,f,g,o; writes h,c (f32 outputs) and
// the h half of the bf16 [h|context] concat buffer for the combine GEMM.
__global__ void __launch_bounds__(256)
k_lstm(const float* gates, const float* bih, const float* bhh,
       const float* c0, float* out_h, float* out_c, __bf16* hcbf) {
  int idx = blockIdx.x * 256 + threadIdx.x;   // 65536
  int b = idx >> 10, h = idx & 1023;
  const float* g = gates + (size_t)b * FOURH;
  float ig = g[h]          + bih[h]          + bhh[h];
  float fg = g[1024 + h]   + bih[1024 + h]   + bhh[1024 + h];
  float gg = g[2048 + h]   + bih[2048 + h]   + bhh[2048 + h];
  float og = g[3072 + h]   + bih[3072 + h]   + bhh[3072 + h];
  float c = sigm(fg) * c0[idx] + sigm(ig) * tanhf(gg);
  float hh = sigm(og) * tanhf(c);
  out_c[idx] = c;
  out_h[idx] = hh;
  hcbf[(size_t)b * H2 + h] = f2bf(hh);
}

// score[b,s] = u2 . enc[s,b,:]   (one wave per (s,b) pair; streams 268 MB)
__global__ void __launch_bounds__(256)
k_score(const float* enc, const float* u2, float* score) {
  const int wave = threadIdx.x >> 5, lane = threadIdx.x & 31;
  const int p = blockIdx.x * 8 + wave;        // 65536 pairs
  const int b = p & 63, s = p >> 6;
  const float* e = enc + ((size_t)s * BB + b) * HH;
  __builtin_prefetch(e + HH, 0, 1);           // next row likely wanted soon
  float acc = 0.f;
#pragma unroll
  for (int i = 0; i < 8; ++i) {
    int k = (i * 32 + lane) * 4;
    float4 ev = *reinterpret_cast<const float4*>(e + k);
    float4 uv = *reinterpret_cast<const float4*>(u2 + k);
    acc += ev.x * uv.x + ev.y * uv.y + ev.z * uv.z + ev.w * uv.w;
  }
#pragma unroll
  for (int m = 16; m >= 1; m >>= 1) acc += __shfl_xor(acc, m, 32);
  if (lane == 0) score[(size_t)b * SS + s] = acc;
}

// row softmax over S=1024 (one block per batch row)
__global__ void __launch_bounds__(256)
k_softmax(const float* score, float* attn_ws, float* attn_out) {
  __shared__ float redmax[8];
  __shared__ float redsum[8];
  const int b = blockIdx.x, t = threadIdx.x;
  const int lane = t & 31, wid = t >> 5;
  const float* srow = score + (size_t)b * SS;
  float v0[4];
  float lm = -3.4e38f;
#pragma unroll
  for (int i = 0; i < 4; ++i) { v0[i] = srow[t + i * 256]; lm = fmaxf(lm, v0[i]); }
#pragma unroll
  for (int o = 16; o >= 1; o >>= 1) lm = fmaxf(lm, __shfl_xor(lm, o, 32));
  if (lane == 0) redmax[wid] = lm;
  __syncthreads();
  float gm = redmax[0];
#pragma unroll
  for (int i = 1; i < 8; ++i) gm = fmaxf(gm, redmax[i]);
  float e[4];
  float ls = 0.f;
#pragma unroll
  for (int i = 0; i < 4; ++i) { e[i] = expf(v0[i] - gm); ls += e[i]; }
#pragma unroll
  for (int o = 16; o >= 1; o >>= 1) ls += __shfl_xor(ls, o, 32);
  if (lane == 0) redsum[wid] = ls;
  __syncthreads();
  float gs = 0.f;
#pragma unroll
  for (int i = 0; i < 8; ++i) gs += redsum[i];
  float inv = 1.f / gs;
#pragma unroll
  for (int i = 0; i < 4; ++i) {
    float a = e[i] * inv;
    attn_ws [(size_t)b * SS + t + i * 256] = a;
    attn_out[(size_t)b * SS + t + i * 256] = a;
  }
}

// context partials: block (b, chunk) accumulates 64 s-steps; 2nd enc sweep
__global__ void __launch_bounds__(256)
k_ctx_part(const float* enc, const float* attn, float* part) {
  const int b = blockIdx.x, chunk = blockIdx.y, t = threadIdx.x;
  __shared__ float aw[64];
  if (t < 64) aw[t] = attn[(size_t)b * SS + chunk * 64 + t];
  __syncthreads();
  const int hbase = t * 4;
  float4 acc = {0.f, 0.f, 0.f, 0.f};
  for (int j = 0; j < 64; ++j) {
    const int s = chunk * 64 + j;
    const float* src = enc + ((size_t)s * BB + b) * HH + hbase;
    if (j + 4 < 64)
      __builtin_prefetch(src + (size_t)4 * BB * HH, 0, 1);
    float4 ev = *reinterpret_cast<const float4*>(src);
    float w = aw[j];
    acc.x += w * ev.x; acc.y += w * ev.y; acc.z += w * ev.z; acc.w += w * ev.w;
  }
  *reinterpret_cast<float4*>(part + (((size_t)chunk * BB + b) * HH + hbase)) = acc;
}

// reduce 16 partials -> context half of the bf16 concat buffer
__global__ void __launch_bounds__(256)
k_ctx_reduce(const float* part, __bf16* hcbf) {
  int idx = blockIdx.x * 256 + threadIdx.x;   // 65536
  int b = idx >> 10, h = idx & 1023;
  float acc = 0.f;
#pragma unroll
  for (int c = 0; c < 16; ++c) acc += part[((size_t)c * BB + b) * HH + h];
  hcbf[(size_t)b * H2 + HH + h] = f2bf(acc);
}

// tmp[b,n] = tanh( [h|ctx][b,:] . Wal[n,:] + bal[n] ), K = 2048
__global__ void __launch_bounds__(32)
k_combine(const __bf16* hcbf, const float* Wal, const float* bal, __bf16* tmpbf) {
  const int lane = threadIdx.x & 31, lo = lane & 15, hi = lane >> 4;
  const int n0 = blockIdx.x * 16;
  const float* wrow = Wal + (size_t)(n0 + lo) * H2;
  v8f acc[4] = {};
  for (int k0 = 0; k0 < H2; k0 += 32) {
    v16bf bt = load_b_row(wrow + k0, hi);
#pragma unroll
    for (int t = 0; t < 4; ++t) {
      v16bf at = load_a_row(hcbf + (size_t)(t * 16 + lo) * H2 + k0, hi);
      acc[t] = wmma_bf16(at, bt, acc[t]);
    }
  }
  float bb = bal[n0 + lo];
#pragma unroll
  for (int t = 0; t < 4; ++t)
#pragma unroll
    for (int r = 0; r < 8; ++r)
      tmpbf[(size_t)(t * 16 + hi * 8 + r) * HH + n0 + lo] =
          f2bf(tanhf(acc[t][r] + bb));
}

// logits[b,n] = tmp[b,:] . Wout[n,:] + bout[n]; streams Wout (206 MB).
// Software-pipelined: tiles for k-step n+1 load while step n's WMMAs issue.
__global__ void __launch_bounds__(32)
k_logits(const __bf16* tmpbf, const float* Wout, const float* bout, float* out) {
  const int lane = threadIdx.x & 31, lo = lane & 15, hi = lane >> 4;
  const int n0 = blockIdx.x * 16;
  const int col = n0 + lo;
  const int jc = col < VV ? col : VV - 1;     // clamp loads for the tail tile
  const float* wrow = Wout + (size_t)jc * HH;

  v8f acc[4] = {};
  v16bf bt = load_b_row(wrow, hi);
  v16bf at[4];
#pragma unroll
  for (int t = 0; t < 4; ++t)
    at[t] = load_a_row(tmpbf + (size_t)(t * 16 + lo) * HH, hi);

  for (int k0 = 32; k0 < HH; k0 += 32) {
    __builtin_prefetch(wrow + k0 + 32, 0, 1);
    v16bf btn = load_b_row(wrow + k0, hi);
    v16bf atn[4];
#pragma unroll
    for (int t = 0; t < 4; ++t)
      atn[t] = load_a_row(tmpbf + (size_t)(t * 16 + lo) * HH + k0, hi);
#pragma unroll
    for (int t = 0; t < 4; ++t)
      acc[t] = wmma_bf16(at[t], bt, acc[t]);
    bt = btn;
#pragma unroll
    for (int t = 0; t < 4; ++t) at[t] = atn[t];
  }
#pragma unroll
  for (int t = 0; t < 4; ++t)
    acc[t] = wmma_bf16(at[t], bt, acc[t]);

  if (col < VV) {
    float bb = bout[col];
#pragma unroll
    for (int t = 0; t < 4; ++t)
#pragma unroll
      for (int r = 0; r < 8; ++r)
        out[(size_t)(t * 16 + hi * 8 + r) * VV + col] = acc[t][r] + bb;
  }
}

// ---------- workspace layout (bytes) ----------
static constexpr size_t OFF_XBF   = 0;                         // 64*1024 bf16
static constexpr size_t OFF_H0BF  = OFF_XBF   + 131072;        // 64*1024 bf16
static constexpr size_t OFF_GATES = OFF_H0BF  + 131072;        // 64*4096 f32
static constexpr size_t OFF_U2    = OFF_GATES + 1048576;       // 1024 f32
static constexpr size_t OFF_HCBF  = OFF_U2    + 4096;          // 64*2048 bf16
static constexpr size_t OFF_SCORE = OFF_HCBF  + 262144;        // 64*1024 f32
static constexpr size_t OFF_ATTN  = OFF_SCORE + 262144;        // 64*1024 f32
static constexpr size_t OFF_PART  = OFF_ATTN  + 262144;        // 16*64*1024 f32
static constexpr size_t OFF_TMPBF = OFF_PART  + 4194304;       // 64*1024 bf16

extern "C" void kernel_launch(void* const* d_in, const int* in_sizes, int n_in,
                              void* d_out, int out_size, void* d_ws, size_t ws_size,
                              hipStream_t stream) {
  const float* enc   = (const float*)d_in[0];
  const int*   seq   = (const int*)  d_in[1];
  const float* h0    = (const float*)d_in[2];
  const float* c0    = (const float*)d_in[3];
  const float* embed = (const float*)d_in[4];
  const float* Wih   = (const float*)d_in[5];
  const float* Whh   = (const float*)d_in[6];
  const float* bih   = (const float*)d_in[7];
  const float* bhh   = (const float*)d_in[8];
  const float* Wa    = (const float*)d_in[9];
  // d_in[10] = ba, mathematically eliminated (softmax shift invariance)
  const float* vvec  = (const float*)d_in[11];
  const float* Wal   = (const float*)d_in[12];
  const float* bal   = (const float*)d_in[13];
  const float* Wout  = (const float*)d_in[14];
  const float* bout  = (const float*)d_in[15];

  char* ws = (char*)d_ws;
  __bf16* xbf   = (__bf16*)(ws + OFF_XBF);
  __bf16* h0bf  = (__bf16*)(ws + OFF_H0BF);
  float*  gates = (float*)(ws + OFF_GATES);
  float*  u2    = (float*)(ws + OFF_U2);
  __bf16* hcbf  = (__bf16*)(ws + OFF_HCBF);
  float*  score = (float*)(ws + OFF_SCORE);
  float*  attn  = (float*)(ws + OFF_ATTN);
  float*  part  = (float*)(ws + OFF_PART);
  __bf16* tmpbf = (__bf16*)(ws + OFF_TMPBF);

  float* out_logits = (float*)d_out;                       // [B, V]
  float* out_h      = out_logits + (size_t)BB * VV;        // [1, B, H]
  float* out_c      = out_h + (size_t)BB * HH;             // [1, B, H]
  float* out_attn   = out_c + (size_t)BB * HH;             // [B, 1, S]

  k_prep   <<<BB, 256, 0, stream>>>(seq, embed, h0, xbf, h0bf);
  k_u2     <<<HH / 256, 256, 0, stream>>>(Wa, vvec, u2);
  k_gates  <<<FOURH / 16, 32, 0, stream>>>(xbf, h0bf, Wih, Whh, gates);
  k_lstm   <<<(BB * HH) / 256, 256, 0, stream>>>(gates, bih, bhh, c0,
                                                 out_h, out_c, hcbf);
  k_score  <<<(SS * BB) / 8, 256, 0, stream>>>(enc, u2, score);
  k_softmax<<<BB, 256, 0, stream>>>(score, attn, out_attn);
  k_ctx_part<<<dim3(BB, 16), 256, 0, stream>>>(enc, attn, part);
  k_ctx_reduce<<<(BB * HH) / 256, 256, 0, stream>>>(part, hcbf);
  k_combine<<<HH / 16, 32, 0, stream>>>(hcbf, Wal, bal, tmpbf);
  k_logits <<<(VV + 15) / 16, 32, 0, stream>>>(tmpbf, Wout, bout, out_logits);
}
